// AoEBlock_11184094839571
// MI455X (gfx1250) — matrix operand order repaired and checked by reading, not compiled, not run on hardware
//
#include <hip/hip_runtime.h>
#include <hip/hip_bf16.h>
#include <math.h>

// ---------------------------------------------------------------------------
// AoE block for MI455X (gfx1250):
//  - 4 GEMMs via V_WMMA_F32_16X16X32_BF16 (fp32 accumulate)
//  - double-buffered GLOBAL_LOAD_ASYNC_TO_LDS_B128 staging (ASYNCcnt)
//  - weights kept as B^T [N,K] so A/B LDS layouts are identical (no packing)
//  - 128x128 macro-tile, 8 waves, each wave 32x64 = 2x4 WMMA tiles
// ---------------------------------------------------------------------------

#define T_TOK   8192          // B * H * W = 8 * 32 * 32
#define C_DIM   768
#define HID     3072
#define ELOW    1536          // NUM_EXPERTS * D_LOW
#define NEXP    8
#define DLOW    192
#define NPIX    1024          // H * W
#define NBATCH  8

typedef __attribute__((ext_vector_type(16))) __bf16 v16bf;
typedef __attribute__((ext_vector_type(8)))  float  v8f;

union FragAB { v16bf v; unsigned int u[8]; };
union FragC  { v8f   v; float        f[8]; };

__device__ __forceinline__ unsigned short f2bf(float f) {
    unsigned int u = __float_as_uint(f);
    unsigned int r = u + 0x7fffu + ((u >> 16) & 1u);   // round-to-nearest-even
    return (unsigned short)(r >> 16);
}
__device__ __forceinline__ float bf2f(unsigned short b) {
    return __uint_as_float(((unsigned int)b) << 16);
}
__device__ __forceinline__ float gelu_exact(float v) {
    return 0.5f * v * (1.0f + erff(v * 0.70710678118654752440f));
}

// issue one per-lane 16B async copy global -> LDS (GV mode, tracked by ASYNCcnt)
__device__ __forceinline__ void async_cp16(unsigned lds_byte_addr, const void* gptr) {
    asm volatile("global_load_async_to_lds_b128 %0, %1, off"
                 :: "v"(lds_byte_addr), "v"((unsigned long long)gptr)
                 : "memory");
}
__device__ __forceinline__ void async_wait0() {
    asm volatile("s_wait_asynccnt 0x0" ::: "memory");
}

// ------------------------------ prep kernels -------------------------------

__global__ void zero_acc_kernel(float* acc) {
    int i = threadIdx.x;
    if (i < 16) acc[i] = 0.0f;
}

// x [B, C, N] fp32 -> xb [T, C] bf16 (token-major)
__global__ void prep_x_kernel(const float* __restrict__ x, unsigned short* __restrict__ xb) {
    size_t idx = (size_t)blockIdx.x * blockDim.x + threadIdx.x;
    size_t total = (size_t)NBATCH * C_DIM * NPIX;
    if (idx >= total) return;
    int n = (int)(idx % NPIX);
    int c = (int)((idx / NPIX) % C_DIM);
    int b = (int)(idx / ((size_t)NPIX * C_DIM));
    xb[((size_t)(b * NPIX + n)) * C_DIM + c] = f2bf(x[idx]);
}

// in [R, Cc] fp32 -> out [Cc, R] bf16
__global__ void prep_transpose_kernel(const float* __restrict__ in, unsigned short* __restrict__ out,
                                      int R, int Cc) {
    size_t idx = (size_t)blockIdx.x * blockDim.x + threadIdx.x;
    size_t total = (size_t)R * Cc;
    if (idx >= total) return;
    int r = (int)(idx / Cc);
    int c = (int)(idx % Cc);
    out[(size_t)c * R + r] = f2bf(in[idx]);
}

// straight fp32 -> bf16 copy
__global__ void prep_copy_kernel(const float* __restrict__ in, unsigned short* __restrict__ out, size_t total) {
    size_t idx = (size_t)blockIdx.x * blockDim.x + threadIdx.x;
    if (idx >= total) return;
    out[idx] = f2bf(in[idx]);
}

// ------------------------------ WMMA GEMM ----------------------------------
// C[M,N] = A[M,K](bf16,row-major) * Bt[N,K](bf16,row-major)^T + bias[N]
// EPI: 0 = store bf16, 1 = GELU then store bf16, 2 = store fp32
// 256 threads = 8 waves. Macro-tile 128x128, BK=32, wave grid 4(M) x 2(N),
// each wave computes a 32x64 patch = 2x4 WMMA 16x16 tiles.
// Double-buffered async global->LDS staging.

template <int EPI>
__global__ __launch_bounds__(256) void gemm_wmma_kernel(
    const unsigned short* __restrict__ A,
    const unsigned short* __restrict__ Bt,
    const float* __restrict__ bias,
    void* __restrict__ Cout,
    int M, int N, int K)
{
    constexpr int BM = 128, BN = 128, BK = 32;
    constexpr int AS = BK + 8;                    // LDS row stride (halfs)
    constexpr int TILE = (BM + BN) * AS;          // halfs per buffer

    __shared__ unsigned short smem[2 * TILE];     // [buf][A rows | B rows][AS]

    const int tid   = threadIdx.x;
    const int lane  = tid & 31;
    const int wave  = tid >> 5;
    const int waveM = wave & 3;                   // 0..3 (32 rows each)
    const int waveN = wave >> 2;                  // 0..1 (64 cols each)
    const int half  = lane >> 4;
    const int l16   = lane & 15;

    const int bM = blockIdx.y * BM;
    const int bN = blockIdx.x * BN;

    // staging mapping: 256 threads; sr = row 0..63, sc = 16B column group
    const int sr = tid >> 2;                      // 0..63
    const int sc = (tid & 3) * 8;                 // 0,8,16,24 (halfs)

    const unsigned smemBase = (unsigned)(size_t)(&smem[0]);   // LDS byte addr

    FragC acc[2][4];
#pragma unroll
    for (int i = 0; i < 2; ++i)
#pragma unroll
        for (int j = 0; j < 4; ++j)
#pragma unroll
            for (int v = 0; v < 8; ++v) acc[i][j].f[v] = 0.0f;

    auto stage = [&](int buf, int k0) {
        unsigned ldsA = smemBase + (unsigned)(buf * TILE) * 2u;
        unsigned ldsB = ldsA + (unsigned)(BM * AS) * 2u;
        const unsigned short* ga0 = A  + (size_t)(bM + sr)      * K + (k0 + sc);
        const unsigned short* ga1 = A  + (size_t)(bM + sr + 64) * K + (k0 + sc);
        const unsigned short* gb0 = Bt + (size_t)(bN + sr)      * K + (k0 + sc);
        const unsigned short* gb1 = Bt + (size_t)(bN + sr + 64) * K + (k0 + sc);
        async_cp16(ldsA + (unsigned)( sr       * AS + sc) * 2u, ga0);
        async_cp16(ldsA + (unsigned)((sr + 64) * AS + sc) * 2u, ga1);
        async_cp16(ldsB + (unsigned)( sr       * AS + sc) * 2u, gb0);
        async_cp16(ldsB + (unsigned)((sr + 64) * AS + sc) * 2u, gb1);
    };

    const int nk = K / BK;
    stage(0, 0);
    async_wait0();
    __syncthreads();

    for (int ki = 0; ki < nk; ++ki) {
        const unsigned short* As_ = &smem[(ki & 1) * TILE];
        const unsigned short* Bs_ = As_ + BM * AS;

        if (ki + 1 < nk) stage((ki + 1) & 1, (ki + 1) * BK);

        // ---- fragments (ISA 7.12.2 bf16 layouts) ----
        FragAB af[2];
#pragma unroll
        for (int tM = 0; tM < 2; ++tM) {
            int r = waveM * 32 + tM * 16 + l16;
            const unsigned short* ap = As_ + r * AS;
#pragma unroll
            for (int p = 0; p < 8; ++p) {
                int kk = (p < 4 ? 2 * p : 2 * p + 8) + half * 8;  // A: K of even elem
                af[tM].u[p] = *(const unsigned int*)(ap + kk);
            }
        }
        FragAB bfr[4];
#pragma unroll
        for (int tN = 0; tN < 4; ++tN) {
            int c = waveN * 64 + tN * 16 + l16;
            const unsigned short* bp = Bs_ + c * AS;
#pragma unroll
            for (int p = 0; p < 8; ++p) {
                int kk = (p < 4 ? 2 * p : 2 * p + 8) + half * 8;  // B: K of even elem
                bfr[tN].u[p] = *(const unsigned int*)(bp + kk);
            }
        }

#pragma unroll
        for (int tM = 0; tM < 2; ++tM)
#pragma unroll
            for (int tN = 0; tN < 4; ++tN)
                acc[tM][tN].v = __builtin_amdgcn_wmma_f32_16x16x32_bf16(
                    false, af[tM].v, false, bfr[tN].v,
                    (short)0, acc[tM][tN].v, false, false);

        async_wait0();      // next tile fully resident in the other buffer
        __syncthreads();    // all waves done reading current buffer
    }

    // ---- epilogue: bias (+GELU) ----
#pragma unroll
    for (int tM = 0; tM < 2; ++tM) {
#pragma unroll
        for (int tN = 0; tN < 4; ++tN) {
            int col = bN + waveN * 64 + tN * 16 + l16;
            float bv = bias ? bias[col] : 0.0f;
#pragma unroll
            for (int v = 0; v < 8; ++v) {
                int row = bM + waveM * 32 + tM * 16 + v + 8 * half;
                float val = acc[tM][tN].f[v] + bv;
                if (EPI == 1) val = gelu_exact(val);
                if (EPI <= 1)
                    ((unsigned short*)Cout)[(size_t)row * N + col] = f2bf(val);
                else
                    ((float*)Cout)[(size_t)row * N + col] = val;
            }
        }
    }
}

// ------------------------------- router ------------------------------------

__global__ __launch_bounds__(256) void router_kernel(
    const unsigned short* __restrict__ feats,    // [T, 1536] bf16
    const float* __restrict__ router_w,          // [192]
    unsigned short* __restrict__ actbuf,         // [T, 1536] bf16
    float* __restrict__ acc)                     // [16]: probsum[8], loadsum[8]
{
    __shared__ float sacc[16];
    const int tid = threadIdx.x, lane = tid & 31, wv = tid >> 5;
    if (tid < 16) sacc[tid] = 0.0f;
    __syncthreads();

    const int t = blockIdx.x * 8 + wv;           // grid = 1024 blocks
    const unsigned short* fr = feats + (size_t)t * ELOW;

    float lg[NEXP];
#pragma unroll
    for (int e = 0; e < NEXP; ++e) {
        float s = 0.0f;
        for (int d = lane; d < DLOW; d += 32)
            s += bf2f(fr[e * DLOW + d]) * router_w[d];
#pragma unroll
        for (int off = 16; off > 0; off >>= 1) s += __shfl_xor(s, off, 32);
        lg[e] = s;
    }

    float m = lg[0];
#pragma unroll
    for (int e = 1; e < NEXP; ++e) m = fmaxf(m, lg[e]);
    float p[NEXP], sum = 0.0f;
#pragma unroll
    for (int e = 0; e < NEXP; ++e) { p[e] = expf(lg[e] - m); sum += p[e]; }
    float inv = 1.0f / sum;
#pragma unroll
    for (int e = 0; e < NEXP; ++e) p[e] *= inv;

    int i0 = 0;
#pragma unroll
    for (int e = 1; e < NEXP; ++e) if (p[e] > p[i0]) i0 = e;
    int i1 = (i0 == 0) ? 1 : 0;
#pragma unroll
    for (int e = 0; e < NEXP; ++e) if (e != i0 && p[e] > p[i1]) i1 = e;
    float gs = p[i0] + p[i1];
    float g0 = p[i0] / gs, g1 = p[i1] / gs;

    if (lane == 0) {
#pragma unroll
        for (int e = 0; e < NEXP; ++e) atomicAdd(&sacc[e], p[e]);
        atomicAdd(&sacc[8 + i0], 1.0f);
        atomicAdd(&sacc[8 + i1], 1.0f);
    }

    unsigned short* ar = actbuf + (size_t)t * ELOW;
#pragma unroll
    for (int e = 0; e < NEXP; ++e) {
        float gate = (e == i0) ? g0 : ((e == i1) ? g1 : 0.0f);
        for (int d = lane; d < DLOW; d += 32) {
            float v = 0.0f;
            if (gate != 0.0f) v = gate * gelu_exact(bf2f(fr[e * DLOW + d]));
            ar[e * DLOW + d] = f2bf(v);
        }
    }

    __syncthreads();
    if (tid < 16) atomicAdd(&acc[tid], sacc[tid]);
}

// ------------------------------ finalize -----------------------------------

__global__ void finalize_kernel(const float* __restrict__ x,
                                const float* __restrict__ shared_o,
                                const float* __restrict__ aoe,
                                float* __restrict__ out)
{
    size_t idx = (size_t)blockIdx.x * blockDim.x + threadIdx.x;
    size_t total = (size_t)NBATCH * C_DIM * NPIX;
    if (idx >= total) return;
    int n = (int)(idx % NPIX);
    int c = (int)((idx / NPIX) % C_DIM);
    int b = (int)(idx / ((size_t)NPIX * C_DIM));
    size_t t = (size_t)(b * NPIX + n);
    out[idx] = x[idx] + shared_o[t * C_DIM + c] + aoe[t * C_DIM + c];
}

__global__ void aux_kernel(const float* __restrict__ acc, float* __restrict__ out_aux) {
    if (threadIdx.x == 0 && blockIdx.x == 0) {
        float s = 0.0f;
        for (int e = 0; e < NEXP; ++e) s += acc[e] * acc[8 + e];
        out_aux[0] = (float)NEXP * s / ((float)T_TOK * (float)T_TOK);
    }
}

// ------------------------------ launcher -----------------------------------

extern "C" void kernel_launch(void* const* d_in, const int* in_sizes, int n_in,
                              void* d_out, int out_size, void* d_ws, size_t ws_size,
                              hipStream_t stream) {
    const float* x        = (const float*)d_in[0];  // [8,768,32,32]
    const float* w1       = (const float*)d_in[1];  // [3072,768]  (= B^T for G1)
    const float* b1       = (const float*)d_in[2];  // [3072]
    const float* w2       = (const float*)d_in[3];  // [768,3072]  (= B^T for G2)
    const float* b2       = (const float*)d_in[4];  // [768]
    const float* w_down   = (const float*)d_in[5];  // [1536,768]  (= B^T for G3)
    const float* router_w = (const float*)d_in[6];  // [1,192]
    const float* w_up     = (const float*)d_in[7];  // [1536,768]  (needs T for G4)
    float* out = (float*)d_out;                     // 6291456 + 1 (aux)

    // -------- workspace layout (bytes, 256-aligned) --------
    char* ws = (char*)d_ws;
    size_t off = 0;
    auto take = [&](size_t bytes) { size_t o = off; off += (bytes + 255) & ~(size_t)255; return o; };
    float*          acc   = (float*)         (ws + take(16 * sizeof(float)));
    unsigned short* xb    = (unsigned short*)(ws + take((size_t)T_TOK * C_DIM * 2));
    unsigned short* w1b   = (unsigned short*)(ws + take((size_t)HID * C_DIM * 2));   // [3072,768]
    unsigned short* w2b   = (unsigned short*)(ws + take((size_t)C_DIM * HID * 2));   // [768,3072]
    unsigned short* wdb   = (unsigned short*)(ws + take((size_t)ELOW * C_DIM * 2));  // [1536,768]
    unsigned short* wub   = (unsigned short*)(ws + take((size_t)C_DIM * ELOW * 2));  // [768,1536]
    unsigned short* hact  = (unsigned short*)(ws + take((size_t)T_TOK * HID * 2));
    unsigned short* feats = (unsigned short*)(ws + take((size_t)T_TOK * ELOW * 2));
    unsigned short* actb  = (unsigned short*)(ws + take((size_t)T_TOK * ELOW * 2));
    float*          shr   = (float*)         (ws + take((size_t)T_TOK * C_DIM * 4));
    float*          aoe   = (float*)         (ws + take((size_t)T_TOK * C_DIM * 4));
    (void)ws_size; (void)in_sizes; (void)n_in; (void)out_size;

    const int TPB = 256;
    auto blocks = [](size_t n, int tpb) { return (unsigned)((n + tpb - 1) / tpb); };

    // 1) zero aux accumulators (harness does not re-poison between replays)
    zero_acc_kernel<<<1, 64, 0, stream>>>(acc);

    // 2) precision conversion (weights stay in native [out,in] = B^T layout)
    prep_x_kernel<<<blocks((size_t)T_TOK * C_DIM, TPB), TPB, 0, stream>>>(x, xb);
    prep_copy_kernel<<<blocks((size_t)HID * C_DIM, TPB), TPB, 0, stream>>>(w1, w1b, (size_t)HID * C_DIM);
    prep_copy_kernel<<<blocks((size_t)C_DIM * HID, TPB), TPB, 0, stream>>>(w2, w2b, (size_t)C_DIM * HID);
    prep_copy_kernel<<<blocks((size_t)ELOW * C_DIM, TPB), TPB, 0, stream>>>(w_down, wdb, (size_t)ELOW * C_DIM);
    prep_transpose_kernel<<<blocks((size_t)ELOW * C_DIM, TPB), TPB, 0, stream>>>(w_up, wub, ELOW, C_DIM);

    // 3) G1: h = GELU(x @ w1^T + b1)      [8192x768]x[3072,768]^T -> bf16
    gemm_wmma_kernel<1><<<dim3(HID / 128, T_TOK / 128), TPB, 0, stream>>>(xb, w1b, b1, hact, T_TOK, HID, C_DIM);

    // 4) G3: feats = x @ w_down^T         [8192x768]x[1536,768]^T -> bf16
    gemm_wmma_kernel<0><<<dim3(ELOW / 128, T_TOK / 128), TPB, 0, stream>>>(xb, wdb, nullptr, feats, T_TOK, ELOW, C_DIM);

    // 5) router: softmax/top2, actbuf = gate*GELU(feats) (zeros for unselected)
    router_kernel<<<T_TOK / 8, TPB, 0, stream>>>(feats, router_w, actb, acc);

    // 6) G2: shared = h @ w2^T + b2       [8192x3072]x[768,3072]^T -> fp32
    gemm_wmma_kernel<2><<<dim3(C_DIM / 128, T_TOK / 128), TPB, 0, stream>>>(hact, w2b, b2, shr, T_TOK, C_DIM, HID);

    // 7) G4: aoe = actbuf @ w_up_flat     [8192x1536]x[768,1536]^T -> fp32
    //    (dense over all experts; zero gates for unselected == sparse top-2)
    gemm_wmma_kernel<2><<<dim3(C_DIM / 128, T_TOK / 128), TPB, 0, stream>>>(actb, wub, nullptr, aoe, T_TOK, C_DIM, ELOW);

    // 8) out = x + shared + aoe (back to [B,C,H,W]); aux loss scalar
    finalize_kernel<<<blocks((size_t)T_TOK * C_DIM, TPB), TPB, 0, stream>>>(x, shr, aoe, out);
    aux_kernel<<<1, 32, 0, stream>>>(acc, out + (size_t)T_TOK * C_DIM);
}
